// CrossAttentionBlock_34840774705253
// MI455X (gfx1250) — matrix operand order repaired and checked by reading, not compiled, not run on hardware
//
#include <hip/hip_runtime.h>
#include <hip/hip_bf16.h>

// ---------------------------------------------------------------------------
// CDNA5 (gfx1250) cross-attention block: Q/K/V projections, flash-style
// masked softmax attention with TDM (tensor_load_to_lds) double-buffered K/V
// staging, output projection, residual + LayerNorm.
// Matmuls (incl. softmax row-sums) run on v_wmma_f32_16x16x32_f16.
// ---------------------------------------------------------------------------

typedef __attribute__((ext_vector_type(16))) _Float16 v16h;
typedef __attribute__((ext_vector_type(8)))  _Float16 v8h;
typedef __attribute__((ext_vector_type(8)))  float    v8f;
typedef __attribute__((ext_vector_type(4)))  unsigned int u32x4;
typedef __attribute__((ext_vector_type(8)))  int      i32x8;
typedef __attribute__((ext_vector_type(4)))  int      i32x4;

#define WMMA_F32_16x16x32_F16(A, B, C) \
  __builtin_amdgcn_wmma_f32_16x16x32_f16(false, (A), false, (B), (short)0, (C), false, false)

// ---- fragment loaders (contiguous k = 16*half + s for A and B, wave32) -------------------------
__device__ inline v16h a_frag_h(const _Float16* __restrict__ row, int half, int k0) {
  v8h lo = *(const v8h*)(row + k0 + half * 16);
  v8h hi = *(const v8h*)(row + k0 + half * 16 + 8);
  v16h a;
#pragma unroll
  for (int i = 0; i < 8; ++i) { a[i] = lo[i]; a[i + 8] = hi[i]; }
  return a;
}
__device__ inline v16h a_frag_f(const float* __restrict__ row, int half, int k0) {
  const float* p0 = row + k0 + half * 16;
  const float* p1 = row + k0 + half * 16 + 8;
  v16h a;
#pragma unroll
  for (int i = 0; i < 8; ++i) { a[i] = (_Float16)p0[i]; a[i + 8] = (_Float16)p1[i]; }
  return a;
}
__device__ inline v16h b_frag_h(const _Float16* __restrict__ p) {
  v8h lo = *(const v8h*)p;
  v8h hi = *(const v8h*)(p + 8);
  v16h b;
#pragma unroll
  for (int i = 0; i < 8; ++i) { b[i] = lo[i]; b[i + 8] = hi[i]; }
  return b;
}

// ---- 16-lane (half-wave) max reduction: xor masks < 16 stay in the half ----
__device__ inline float rowmax16(float v) {
#pragma unroll
  for (int m = 1; m < 16; m <<= 1) v = fmaxf(v, __shfl_xor(v, m, 32));
  return v;
}

// ---------------------------------------------------------------------------
// TDM: issue one 2D tensor_load_to_lds (f16 elements). D# per ISA ch.8:
//  group0: count=1 | lds_addr | global_addr(57b) | type=2
//  group1: data_size=1 (2B); tensor dims == tile dims (no OOB); stride0 in
//          element units; tile_dim2=0 -> 2D. Remaining groups zero.
// Tracked by TENSORcnt (in-order per wave). This toolchain exposes the
// 6-arg builtin form: (u32x4, i32x8, i32x4, i32x4, i32x8, i32 cpol).
// ---------------------------------------------------------------------------
__device__ inline void tdm_load_2d(unsigned lds_off, unsigned long long gaddr,
                                   unsigned tile_d0, unsigned tile_d1,
                                   unsigned long long stride0) {
  u32x4 g0;
  g0[0] = 1u;                                            // count=1, user mode
  g0[1] = lds_off;                                       // LDS byte address
  g0[2] = (unsigned)(gaddr & 0xffffffffu);               // global_addr[31:0]
  g0[3] = (unsigned)((gaddr >> 32) & 0x1ffffffu)         // global_addr[56:32]
          | (2u << 30);                                  // type=2 ("image")
  i32x8 g1;
  g1[0] = (int)(1u << 16);                               // data_size=1 (2B)
  g1[1] = (int)((tile_d0 & 0xffffu) << 16);              // tensor_dim0[15:0]
  g1[2] = (int)((tile_d0 >> 16) | ((tile_d1 & 0xffffu) << 16)); // td0[31:16]|td1[15:0]
  g1[3] = (int)((tile_d1 >> 16) | (tile_d0 << 16));      // td1[31:16]|tile_dim0
  g1[4] = (int)(tile_d1 & 0xffffu);                      // tile_dim1, tile_dim2=0
  g1[5] = (int)(stride0 & 0xffffffffu);                  // dim0_stride[31:0]
  g1[6] = (int)((stride0 >> 32) & 0xffffu);              // dim0_stride[47:32]
  g1[7] = 0;
  const i32x4 gz4 = {0, 0, 0, 0};
  const i32x8 gz8 = {0, 0, 0, 0, 0, 0, 0, 0};
  __builtin_amdgcn_tensor_load_to_lds(g0, g1, gz4, gz4, gz8, 0);
}

// ---------------------------------------------------------------------------
// One-shot f32 -> f16 conversion (weights).
// ---------------------------------------------------------------------------
__global__ __launch_bounds__(256) void cvt_f16_kernel(
    const float* __restrict__ in, _Float16* __restrict__ out, int n) {
  const int i = (blockIdx.x * 256 + threadIdx.x) * 4;
  if (i + 3 < n) {
#pragma unroll
    for (int j = 0; j < 4; ++j) out[i + j] = (_Float16)in[i + j];
  }
}

// ---------------------------------------------------------------------------
// Projection GEMM: C[M][N] = A[M][K] @ Wh[N][K]^T + bias   (Wh already f16).
// One wave computes a 16x64 tile. MODE 0: f16 [M][N]; MODE 1: f16 Vt layout
// [b][h][d][s]; MODE 2: f32 [M][N]. A_IS_HALF: A operand f16 vs f32.
// ---------------------------------------------------------------------------
template <int MODE, int A_IS_HALF>
__global__ __launch_bounds__(32) void proj_gemm(
    const void* __restrict__ Av, const _Float16* __restrict__ Wh,
    const float* __restrict__ bias, void* __restrict__ out,
    int M, int K, int N, int S, int HD) {
  const int lane = threadIdx.x;
  const int half = lane >> 4;
  const int l16  = lane & 15;
  const int row0 = blockIdx.x * 16;
  const int col0 = blockIdx.y * 64;

  v8f acc[4] = {};
  const float*    Af = (const float*)Av;
  const _Float16* Ah = (const _Float16*)Av;

  for (int k0 = 0; k0 < K; k0 += 32) {
    v16h a;
    if (A_IS_HALF) a = a_frag_h(Ah + (size_t)(row0 + l16) * K, half, k0);
    else           a = a_frag_f(Af + (size_t)(row0 + l16) * K, half, k0);
#pragma unroll
    for (int j = 0; j < 4; ++j) {
      const _Float16* wp = Wh + (size_t)(col0 + j * 16 + l16) * K + k0 + half * 16;
      v16h b = b_frag_h(wp);
      acc[j] = WMMA_F32_16x16x32_F16(a, b, acc[j]);
    }
  }

#pragma unroll
  for (int j = 0; j < 4; ++j) {
    const int col = col0 + j * 16 + l16;
    const float bv = bias[col];
#pragma unroll
    for (int r = 0; r < 8; ++r) {
      const int row = row0 + half * 8 + r;
      const float val = acc[j][r] + bv;
      if (MODE == 2) {
        ((float*)out)[(size_t)row * N + col] = val;
      } else if (MODE == 0) {
        ((_Float16*)out)[(size_t)row * N + col] = (_Float16)val;
      } else {  // MODE 1: Vt[b][h][d][s]
        const int b = row / S, s = row % S;
        const int h = col / HD, d = col % HD;
        const int H = N / HD;
        ((_Float16*)out)[(((size_t)(b * H + h) * HD + d) * S) + s] = (_Float16)val;
      }
    }
  }
}

// ---------------------------------------------------------------------------
// Attention: one wave per (16 query rows, head, batch). K/V tiles for each
// 32-kv chunk are staged into LDS by the Tensor Data Mover, double-buffered:
// chunk c+1's DMA overlaps chunk c's 9 WMMAs (4 scores + 1 rowsum + 4 P@V).
// Row sums via D = P @ ones(32x16) on the matrix core. P transposes
// C-layout -> A-layout through a padded LDS tile (single-wave workgroup).
// ---------------------------------------------------------------------------
__global__ __launch_bounds__(32) void attn_kernel(
    const _Float16* __restrict__ Qp, const _Float16* __restrict__ Kp,
    const _Float16* __restrict__ Vt, const int* __restrict__ mask,
    _Float16* __restrict__ AO, int SQ, int SKV, int E, int H) {
  __shared__ __align__(16) _Float16 Kt[2][32][64];   // kv-row major, 4KB/buf
  __shared__ __align__(16) _Float16 Vtl[2][64][32];  // d-row major,  4KB/buf
  __shared__ __align__(16) _Float16 Pl[16][40];      // padded P transpose tile

  const int lane = threadIdx.x;
  const int half = lane >> 4;
  const int l16  = lane & 15;
  const int q0 = blockIdx.x * 16;
  const int h  = blockIdx.y;
  const int b  = blockIdx.z;
  const int HD = E / H;  // 64
  const float scale = 0.125f;  // 1/sqrt(64)

  const _Float16* qrow = Qp + (size_t)(b * SQ + q0 + l16) * E + h * HD;
  const v16h aq0 = a_frag_h(qrow, half, 0);
  const v16h aq1 = a_frag_h(qrow, half, 32);

  v16h onesb;
#pragma unroll
  for (int i = 0; i < 16; ++i) onesb[i] = (_Float16)1.0f;

  const int*      mrow  = mask + (size_t)b * SKV;
  const _Float16* kbase = Kp + (size_t)b * SKV * E + h * HD;     // [kv][d] stride E
  const _Float16* vbase = Vt + (size_t)(b * H + h) * HD * SKV;   // [d][kv] stride SKV

  const unsigned ldsK[2] = { (unsigned)(size_t)&Kt[0][0][0],  (unsigned)(size_t)&Kt[1][0][0] };
  const unsigned ldsV[2] = { (unsigned)(size_t)&Vtl[0][0][0], (unsigned)(size_t)&Vtl[1][0][0] };

  const int NCH = SKV / 32;
  // prefetch chunk 0 into buffer 0
  tdm_load_2d(ldsK[0], (unsigned long long)(size_t)kbase, 64u, 32u, (unsigned long long)E);
  tdm_load_2d(ldsV[0], (unsigned long long)(size_t)vbase, 32u, 64u, (unsigned long long)SKV);

  v8f o[4] = {};
  float rm[8], rl[8];
#pragma unroll
  for (int r = 0; r < 8; ++r) { rm[r] = -1e30f; rl[r] = 0.0f; }

  for (int c = 0; c < NCH; ++c) {
    const int buf = c & 1;
    const int kv0 = c * 32;
    if (c + 1 < NCH) {  // issue next chunk's DMA, then wait for current buffer
      const int nb = (c + 1) & 1;
      tdm_load_2d(ldsK[nb],
                  (unsigned long long)(size_t)(kbase + (size_t)(kv0 + 32) * E),
                  64u, 32u, (unsigned long long)E);
      tdm_load_2d(ldsV[nb],
                  (unsigned long long)(size_t)(vbase + (size_t)(kv0 + 32)),
                  32u, 64u, (unsigned long long)SKV);
      __builtin_amdgcn_s_wait_tensorcnt(2);   // in-order: current buffer done
    } else {
      __builtin_amdgcn_s_wait_tensorcnt(0);
    }
    __asm__ volatile("" ::: "memory");        // pin LDS reads after the wait

    // ---- scores: S[16x32] = Q(16x64) . K^T, two 16-col tiles (K from LDS) --
    v8f s[2] = {};
#pragma unroll
    for (int j = 0; j < 2; ++j) {
      const _Float16* kp = &Kt[buf][j * 16 + l16][0];
      v16h bk0 = b_frag_h(kp + half * 16);
      v16h bk1 = b_frag_h(kp + 32 + half * 16);
      s[j] = WMMA_F32_16x16x32_F16(aq0, bk0, s[j]);
      s[j] = WMMA_F32_16x16x32_F16(aq1, bk1, s[j]);
    }
    // scale + mask (column index = lane&15 within each tile)
#pragma unroll
    for (int j = 0; j < 2; ++j) {
      const bool keep = mrow[kv0 + j * 16 + l16] != 0;
#pragma unroll
      for (int r = 0; r < 8; ++r)
        s[j][r] = keep ? s[j][r] * scale : -1e30f;
    }
    // online softmax max + exp (rows live per-VGPR per-half)
    float alpha[8];
    v8f p0, p1;
#pragma unroll
    for (int r = 0; r < 8; ++r) {
      float cm = rowmax16(fmaxf(s[0][r], s[1][r]));
      float nm = fmaxf(rm[r], cm);
      alpha[r] = __expf(rm[r] - nm);
      rm[r] = nm;
      p0[r] = __expf(s[0][r] - nm);
      p1[r] = __expf(s[1][r] - nm);
    }
    // ---- transpose P (C-layout f32) -> A-layout f16 via LDS ----
    __syncthreads();  // WAR vs previous iteration's A-frag reads
#pragma unroll
    for (int r = 0; r < 8; ++r) {
      Pl[half * 8 + r][l16]      = (_Float16)p0[r];
      Pl[half * 8 + r][16 + l16] = (_Float16)p1[r];
    }
    __syncthreads();
    const v16h ap = a_frag_h(&Pl[l16][0], half, 0);
    // ---- row sums on the matrix core: D = P @ ones(32x16) ----
    v8f ps = {};
    ps = WMMA_F32_16x16x32_F16(ap, onesb, ps);
#pragma unroll
    for (int r = 0; r < 8; ++r) rl[r] = rl[r] * alpha[r] + ps[r];
    // ---- O = O*alpha + P @ V(32x64), V from LDS ----
#pragma unroll
    for (int j = 0; j < 4; ++j) {
#pragma unroll
      for (int r = 0; r < 8; ++r) o[j][r] *= alpha[r];
      v16h bv = b_frag_h(&Vtl[buf][j * 16 + l16][half * 16]);
      o[j] = WMMA_F32_16x16x32_F16(ap, bv, o[j]);
    }
  }

  // normalize and store f16 attention output [b*SQ+row][h*HD + d]
#pragma unroll
  for (int j = 0; j < 4; ++j) {
#pragma unroll
    for (int r = 0; r < 8; ++r) {
      const int row = q0 + half * 8 + r;
      const int d   = j * 16 + l16;
      AO[(size_t)(b * SQ + row) * E + h * HD + d] = (_Float16)(o[j][r] / rl[r]);
    }
  }
}

// ---------------------------------------------------------------------------
// Residual + LayerNorm: one 256-thread block per row of E=1024.
// ---------------------------------------------------------------------------
__global__ __launch_bounds__(256) void ln_kernel(
    const float* __restrict__ q, const float* __restrict__ po,
    const float* __restrict__ gamma, const float* __restrict__ beta,
    float* __restrict__ out, int E) {
  __shared__ float sh1[8], sh2[8];
  const int row = blockIdx.x;
  const int tid = threadIdx.x;
  const float* xq = q + (size_t)row * E;
  const float* xp = po + (size_t)row * E;

  float vals[4];
  float s1 = 0.0f, s2 = 0.0f;
#pragma unroll
  for (int i = 0; i < 4; ++i) {
    const int c = tid + i * 256;
    const float x = xq[c] + xp[c];
    vals[i] = x;
    s1 += x;
    s2 += x * x;
  }
#pragma unroll
  for (int m = 1; m < 32; m <<= 1) {
    s1 += __shfl_xor(s1, m, 32);
    s2 += __shfl_xor(s2, m, 32);
  }
  const int w = tid >> 5;
  if ((tid & 31) == 0) { sh1[w] = s1; sh2[w] = s2; }
  __syncthreads();
  float t1 = 0.0f, t2 = 0.0f;
#pragma unroll
  for (int i = 0; i < 8; ++i) { t1 += sh1[i]; t2 += sh2[i]; }
  const float mean = t1 / (float)E;
  const float var  = t2 / (float)E - mean * mean;
  const float rstd = rsqrtf(var + 1e-5f);
#pragma unroll
  for (int i = 0; i < 4; ++i) {
    const int c = tid + i * 256;
    out[(size_t)row * E + c] = (vals[i] - mean) * rstd * gamma[c] + beta[c];
  }
}

// ---------------------------------------------------------------------------
extern "C" void kernel_launch(void* const* d_in, const int* in_sizes, int n_in,
                              void* d_out, int out_size, void* d_ws, size_t ws_size,
                              hipStream_t stream) {
  (void)in_sizes; (void)n_in; (void)out_size; (void)ws_size;
  constexpr int B = 2, SQ = 1024, SKV = 4096, E = 1024, H = 16, HD = 64;

  const float* query = (const float*)d_in[0];
  const float* kv    = (const float*)d_in[1];
  const int*   mask  = (const int*)d_in[2];
  const float* Wq    = (const float*)d_in[3];
  const float* bq    = (const float*)d_in[4];
  const float* Wk    = (const float*)d_in[5];
  const float* bk    = (const float*)d_in[6];
  const float* Wv    = (const float*)d_in[7];
  const float* bv    = (const float*)d_in[8];
  const float* Wo    = (const float*)d_in[9];
  const float* bo    = (const float*)d_in[10];
  const float* lng   = (const float*)d_in[11];
  const float* lnb   = (const float*)d_in[12];

  // workspace carve-up (56 MiB total, all 16B-aligned chunks)
  char* ws = (char*)d_ws;
  _Float16* Qp  = (_Float16*)ws;  ws += (size_t)B * SQ  * E * sizeof(_Float16);  //  4 MiB
  _Float16* Kp  = (_Float16*)ws;  ws += (size_t)B * SKV * E * sizeof(_Float16);  // 16 MiB
  _Float16* Vt  = (_Float16*)ws;  ws += (size_t)B * SKV * E * sizeof(_Float16);  // 16 MiB
  _Float16* AO  = (_Float16*)ws;  ws += (size_t)B * SQ  * E * sizeof(_Float16);  //  4 MiB
  float*    PO  = (float*)ws;     ws += (size_t)B * SQ  * E * sizeof(float);     //  8 MiB
  _Float16* Wqh = (_Float16*)ws;  ws += (size_t)E * E * sizeof(_Float16);        //  2 MiB
  _Float16* Wkh = (_Float16*)ws;  ws += (size_t)E * E * sizeof(_Float16);        //  2 MiB
  _Float16* Wvh = (_Float16*)ws;  ws += (size_t)E * E * sizeof(_Float16);        //  2 MiB
  _Float16* Woh = (_Float16*)ws;                                                 //  2 MiB

  const dim3 blk(32);
  // one-shot weight conversion f32 -> f16 (E*E = 1M elems, 4/thread)
  const int nW = E * E;
  const dim3 cgrid(nW / (256 * 4));
  cvt_f16_kernel<<<cgrid, dim3(256), 0, stream>>>(Wq, Wqh, nW);
  cvt_f16_kernel<<<cgrid, dim3(256), 0, stream>>>(Wk, Wkh, nW);
  cvt_f16_kernel<<<cgrid, dim3(256), 0, stream>>>(Wv, Wvh, nW);
  cvt_f16_kernel<<<cgrid, dim3(256), 0, stream>>>(Wo, Woh, nW);

  // Q / K projections -> f16 row-major
  proj_gemm<0, 0><<<dim3(B * SQ  / 16, E / 64), blk, 0, stream>>>(query, Wqh, bq, Qp, B * SQ,  E, E, SQ,  HD);
  proj_gemm<0, 0><<<dim3(B * SKV / 16, E / 64), blk, 0, stream>>>(kv,    Wkh, bk, Kp, B * SKV, E, E, SKV, HD);
  // V projection -> f16 transposed per head: Vt[b][h][d][skv]
  proj_gemm<1, 0><<<dim3(B * SKV / 16, E / 64), blk, 0, stream>>>(kv,    Wvh, bv, Vt, B * SKV, E, E, SKV, HD);
  // masked flash attention (TDM-staged K/V) -> AO f16 [b*SQ][E]
  attn_kernel<<<dim3(SQ / 16, H, B), blk, 0, stream>>>(Qp, Kp, Vt, mask, AO, SQ, SKV, E, H);
  // output projection (f16 A) -> PO f32
  proj_gemm<2, 1><<<dim3(B * SQ / 16, E / 64), blk, 0, stream>>>(AO, Woh, bo, PO, B * SQ, E, E, SQ, HD);
  // residual + LayerNorm -> d_out f32
  ln_kernel<<<dim3(B * SQ), dim3(256), 0, stream>>>(query, PO, lng, lnb, (float*)d_out, E);
}
